// CRF_58841051955732
// MI455X (gfx1250) — matrix operand order repaired and checked by reading, not compiled, not run on hardware
//
#include <hip/hip_runtime.h>
#include <hip/hip_bf16.h>

// Problem constants (match reference)
#define BB 256
#define SS 1024
#define TT 64

#define LOG2E 1.4426950408889634f
#define LN2   0.6931471805599453f

typedef __attribute__((ext_vector_type(8)))  float  v8f;
typedef __attribute__((ext_vector_type(16))) __bf16 v16bf;

union V16 { unsigned u[8]; v16bf v; };

// f32 pair -> packed bf16 dword. Round-half-up bias then one v_perm_b32
// selecting the two high halves: 3 VALU total per pair.
__device__ __forceinline__ unsigned pack_bf16(float lo, float hi) {
  union { float f; unsigned u; } a, b;
  a.f = lo; b.f = hi;
  // result = { (hi+0x8000)[31:16], (lo+0x8000)[31:16] }
  return __builtin_amdgcn_perm(b.u + 0x8000u, a.u + 0x8000u, 0x07060302u);
}

// exchange with partner lane (lane ^ 16) via ds_swizzle (group-of-32, xor=16)
__device__ __forceinline__ float swz_xor16(float x) {
  union { float f; int i; } a, r;
  a.f = x;
  r.i = __builtin_amdgcn_ds_swizzle(a.i, 0x401F); // and=0x1f, or=0, xor=16
  return r.f;
}

// bare-metal v_exp_f32 / v_log_f32 (base-2, no OCML fixup sequences)
__device__ __forceinline__ float fexp2(float x) { return __builtin_amdgcn_exp2f(x); }
__device__ __forceinline__ float flog2(float x) { return __builtin_amdgcn_logf(x);  }

// ---------------------------------------------------------------------------
// Forward (log-partition) recurrence, entirely in base-2.
// Grid: 16 blocks (16 batches each), 128 threads = 4 waves.
// Wave wv owns j-tile [16*wv, 16*wv+16).  WMMA orientation: D = E^T x P,
//   A (const) = exp(trans)^T  : M=j, K=i   (bf16, 16 VGPRs per wave)
//   B (per step) = p          : K=i, N=b   (bf16, rebuilt from LDS)
//   C/D = s / g-hat           : M=j, N=b   (f32) -> lane&15 == batch
// g-hat = (alpha - Csum2*ln2) * log2e ; p = exp2(g-hat) ; s = p x E
// g-hat_new = em*log2e + log2(s) - delta2 ; Csum2 += delta2 (stale-exact max)
// ---------------------------------------------------------------------------
__global__ __launch_bounds__(128) void crf_forward_kernel(
    const float* __restrict__ em, const int* __restrict__ mask,
    const float* __restrict__ trans, const float* __restrict__ startt,
    const float* __restrict__ endt, float* __restrict__ logZ)
{
  __shared__ unsigned Pbuf[2][16][32];   // packed bf16 p pairs: [buf][batch][i/2]
  __shared__ float    Mpart[2][4][16];   // per-wave partial max: [buf][wave][batch]
  __shared__ float    Red[4][16];        // final-reduction scratch

  const int lane  = threadIdx.x & 31;
  const int wv    = threadIdx.x >> 5;    // j-tile index 0..3
  const int bt    = lane & 15;           // batch-in-tile (N role)
  const int h     = lane >> 4;           // lane half
  const int bglob = blockIdx.x * 16 + bt;

  // ---- constant A = exp(transitions)^T for this wave's j-tile -------------
  // A 16-bit 16x32 layout: lane: M = j = 16*wv + (lane&15);
  // VGPR v, half h: K = 32*kc + 16*(v>>2) + 8*h + 2*(v&3) (pair +1)
  V16 A[2];
  const int jA = 16 * wv + bt;
  #pragma unroll
  for (int kc = 0; kc < 2; ++kc) {
    #pragma unroll
    for (int v = 0; v < 8; ++v) {
      int i0 = 32 * kc + 16 * (v >> 2) + 8 * h + 2 * (v & 3);
      float e0 = fexp2(trans[i0 * TT + jA] * LOG2E);
      float e1 = fexp2(trans[(i0 + 1) * TT + jA] * LOG2E);
      A[kc].u[v] = pack_bf16(e0, e1);
    }
  }

  // ---- g-hat_0 in C layout: reg r -> j = 16*wv + 8*h + r ------------------
  const int   jbase = 16 * wv + 8 * h;
  const float* emrow = em + (size_t)bglob * SS * TT;
  float gh[8];
  #pragma unroll
  for (int r = 0; r < 8; ++r)
    gh[r] = (startt[jbase + r] + emrow[jbase + r]) * LOG2E;
  float Csum2 = 0.f;

  // publish p(g-hat_0) and mu_0 into buffer 0
  {
    float ex[8];
    #pragma unroll
    for (int r = 0; r < 8; ++r) ex[r] = fexp2(gh[r]);
    #pragma unroll
    for (int q = 0; q < 4; ++q)
      Pbuf[0][bt][8 * wv + 4 * h + q] = pack_bf16(ex[2 * q], ex[2 * q + 1]);
    float mm = gh[0];
    #pragma unroll
    for (int r = 1; r < 8; ++r) mm = fmaxf(mm, gh[r]);
    mm = fmaxf(mm, swz_xor16(mm));
    if (h == 0) Mpart[0][wv][bt] = mm;
  }
  __syncthreads();

  // ---- recurrence ---------------------------------------------------------
  for (int step = 1; step < SS; ++step) {
    const int rb = (step - 1) & 1;
    const int wb = step & 1;

    // shift for this step (stale-by-one exact max: safe, telescopes exactly)
    float delta2 = fmaxf(fmaxf(Mpart[rb][0][bt], Mpart[rb][1][bt]),
                         fmaxf(Mpart[rb][2][bt], Mpart[rb][3][bt]));
    Csum2 += delta2;

    // B operand (p) from LDS: reg (kc,v): K = 32*kc + 16*(v>>2) + 8*h + 2*(v&3)
    V16 Bm[2];
    #pragma unroll
    for (int kc = 0; kc < 2; ++kc)
      #pragma unroll
      for (int v = 0; v < 8; ++v)
        Bm[kc].u[v] = Pbuf[rb][bt][16 * kc + 8 * (v >> 2) + 4 * h + (v & 3)];

    // emissions for this step: 8 consecutive floats (this lane's batch & j's)
    const float* ep = emrow + (size_t)step * TT + jbase;
    float4 e0 = *reinterpret_cast<const float4*>(ep);
    float4 e1 = *reinterpret_cast<const float4*>(ep + 4);
    float ev[8] = {e0.x, e0.y, e0.z, e0.w, e1.x, e1.y, e1.z, e1.w};
    if (step + 8 < SS)
      __builtin_prefetch(emrow + (size_t)(step + 8) * TT + jbase, 0, 1);

    const bool mok = mask[bglob * SS + step] != 0;

    // s = E^T x p  (two chained bf16 WMMAs, f32 accumulation)
    v8f acc = {0.f, 0.f, 0.f, 0.f, 0.f, 0.f, 0.f, 0.f};
    acc = __builtin_amdgcn_wmma_f32_16x16x32_bf16(
        false, A[0].v, false, Bm[0].v, (short)0, acc, false, false);
    acc = __builtin_amdgcn_wmma_f32_16x16x32_bf16(
        false, A[1].v, false, Bm[1].v, (short)0, acc, false, false);

    // g-hat update (masked): gn = em*log2e + log2(s) - delta2
    #pragma unroll
    for (int r = 0; r < 8; ++r) {
      float gn = fmaf(ev[r], LOG2E, flog2(acc[r]) - delta2);
      float go = gh[r] - delta2;
      gh[r] = mok ? gn : go;
    }

    // publish p and mu into write buffer
    float ex[8];
    #pragma unroll
    for (int r = 0; r < 8; ++r) ex[r] = fexp2(gh[r]);
    #pragma unroll
    for (int q = 0; q < 4; ++q)
      Pbuf[wb][bt][8 * wv + 4 * h + q] = pack_bf16(ex[2 * q], ex[2 * q + 1]);
    float mm = gh[0];
    #pragma unroll
    for (int r = 1; r < 8; ++r) mm = fmaxf(mm, gh[r]);
    mm = fmaxf(mm, swz_xor16(mm));
    if (h == 0) Mpart[wb][wv][bt] = mm;

    __syncthreads();
  }

  // ---- logZ[b] = ln2 * (Csum2 + log2sumexp2_j(g-hat + end*log2e)) ---------
  float vv[8];
  #pragma unroll
  for (int r = 0; r < 8; ++r) vv[r] = fmaf(endt[jbase + r], LOG2E, gh[r]);
  float mm = vv[0];
  #pragma unroll
  for (int r = 1; r < 8; ++r) mm = fmaxf(mm, vv[r]);
  mm = fmaxf(mm, swz_xor16(mm));
  if (h == 0) Red[wv][bt] = mm;
  __syncthreads();
  float M = fmaxf(fmaxf(Red[0][bt], Red[1][bt]), fmaxf(Red[2][bt], Red[3][bt]));
  float ssum = 0.f;
  #pragma unroll
  for (int r = 0; r < 8; ++r) ssum += fexp2(vv[r] - M);
  ssum += swz_xor16(ssum);
  __syncthreads();
  if (h == 0) Red[wv][bt] = ssum;
  __syncthreads();
  if (threadIdx.x < 16) {
    float tot = Red[0][bt] + Red[1][bt] + Red[2][bt] + Red[3][bt];
    logZ[bglob] = LN2 * (Csum2 + M + flog2(tot));
  }
}

// ---------------------------------------------------------------------------
// Gold path score: one block per batch, deterministic tree reduction.
// ---------------------------------------------------------------------------
__global__ __launch_bounds__(256) void crf_score_kernel(
    const float* __restrict__ em, const int* __restrict__ tags,
    const int* __restrict__ mask, const float* __restrict__ trans,
    const float* __restrict__ startt, const float* __restrict__ endt,
    float* __restrict__ score)
{
  __shared__ float redf[256];
  __shared__ int   redi[256];
  const int b = blockIdx.x, tid = threadIdx.x;
  const int*   tb = tags + b * SS;
  const int*   mb = mask + b * SS;
  const float* eb = em + (size_t)b * SS * TT;

  float acc = 0.f; int mcnt = 0;
  for (int s = tid; s < SS; s += 256) {
    int t = tb[s], m = mb[s];
    mcnt += (m != 0);
    if (s > 0 && m != 0)
      acc += trans[tb[s - 1] * TT + t] + eb[(size_t)s * TT + t];
  }
  redf[tid] = acc; redi[tid] = mcnt;
  __syncthreads();
  for (int off = 128; off > 0; off >>= 1) {
    if (tid < off) { redf[tid] += redf[tid + off]; redi[tid] += redi[tid + off]; }
    __syncthreads();
  }
  if (tid == 0) {
    int last = redi[0] - 1;
    int t0 = tb[0];
    score[b] = redf[0] + startt[t0] + eb[t0] + endt[tb[last]];
  }
}

// ---------------------------------------------------------------------------
// Final: mean(logZ - score) over 256 batches (deterministic).
// ---------------------------------------------------------------------------
__global__ __launch_bounds__(256) void crf_final_kernel(
    const float* __restrict__ logZ, const float* __restrict__ score,
    float* __restrict__ out)
{
  __shared__ float red[256];
  int tid = threadIdx.x;
  red[tid] = logZ[tid] - score[tid];
  __syncthreads();
  for (int off = 128; off > 0; off >>= 1) {
    if (tid < off) red[tid] += red[tid + off];
    __syncthreads();
  }
  if (tid == 0) out[0] = red[0] * (1.0f / (float)BB);
}

extern "C" void kernel_launch(void* const* d_in, const int* in_sizes, int n_in,
                              void* d_out, int out_size, void* d_ws, size_t ws_size,
                              hipStream_t stream) {
  const float* emissions = (const float*)d_in[0];
  const int*   tags      = (const int*)d_in[1];
  const int*   mask      = (const int*)d_in[2];
  const float* trans     = (const float*)d_in[3];
  const float* startt    = (const float*)d_in[4];
  const float* endt      = (const float*)d_in[5];
  float* out = (float*)d_out;

  float* logZ  = (float*)d_ws;        // BB floats
  float* score = logZ + BB;           // BB floats

  crf_forward_kernel<<<dim3(BB / 16), dim3(128), 0, stream>>>(
      emissions, mask, trans, startt, endt, logZ);
  crf_score_kernel<<<dim3(BB), dim3(256), 0, stream>>>(
      emissions, tags, mask, trans, startt, endt, score);
  crf_final_kernel<<<dim3(1), dim3(256), 0, stream>>>(logZ, score, out);
}
